// CompletionNetSigMask_53446573032311
// MI455X (gfx1250) — matrix operand order repaired and checked by reading, not compiled, not run on hardware
//
#include <hip/hip_runtime.h>
#include <hip/hip_bf16.h>
#include <math.h>

#define EPS_BN 1e-5f

typedef __attribute__((ext_vector_type(16))) _Float16 v16h;
typedef __attribute__((ext_vector_type(8)))  _Float16 v8h;
typedef __attribute__((ext_vector_type(8)))  float    v8f;

static __device__ __forceinline__ float elu_f(float y) {
  return y > 0.f ? y : (__expf(y) - 1.f);
}

// ---------------------------------------------------------------------------
// Weight convert: W (K, Cin, Cout) f32 -> Wt (K, Cout, 32) f16, transposed so a
// WMMA B-fragment lane load is two contiguous v8h (k 8*hi.. and 16+8*hi..). K-dim padded to 32 with 0.
// Grid: (tiles over Cout*32, K)
// ---------------------------------------------------------------------------
__global__ void convert_w_kernel(const float* __restrict__ W, _Float16* __restrict__ Wt,
                                 int Cin, int Cout) {
  int k   = blockIdx.y;
  int rem = blockIdx.x * blockDim.x + threadIdx.x;
  if (rem >= Cout * 32) return;
  int n = rem >> 5;   // output channel
  int c = rem & 31;   // input channel (K dim of WMMA)
  Wt[k * (Cout * 32) + rem] = (c < Cin) ? (_Float16)W[(k * Cin + c) * Cout + n]
                                        : (_Float16)0.f;
}

__global__ void zero_kernel(float* __restrict__ p, int n) {
  int i = blockIdx.x * blockDim.x + threadIdx.x;
  if (i < n) p[i] = 0.f;
}

// out rows [0, n]: bias for real rows, zero for dump row n.  C = 1<<cshift
__global__ void init_bias_kernel(float* __restrict__ out, const float* __restrict__ b,
                                 int n, int cshift) {
  int idx = blockIdx.x * blockDim.x + threadIdx.x;
  if (idx >= ((n + 1) << cshift)) return;
  int i = idx >> cshift, c = idx & ((1 << cshift) - 1);
  out[idx] = (i < n) ? b[c] : 0.f;
}

// out = bias + act_f16 @ Wc   (central 1x1 term of the sparse conv)
__global__ void init_central_kernel(float* __restrict__ out, const float* __restrict__ b,
                                    const _Float16* __restrict__ act, int Cin,
                                    const float* __restrict__ Wc, int n, int cshift) {
  int idx = blockIdx.x * blockDim.x + threadIdx.x;
  if (idx >= ((n + 1) << cshift)) return;
  int C = 1 << cshift;
  int i = idx >> cshift, c = idx & (C - 1);
  float s = 0.f;
  if (i < n) {
    s = b[c];
    const _Float16* row = act + (long)i * Cin;
    for (int j = 0; j < Cin; ++j) s += (float)row[j] * Wc[j * C + c];
  }
  out[idx] = s;
}

// ---------------------------------------------------------------------------
// Layer A (Cin = 1 -> 16): scalar path, WMMA would waste 31/32 of the MACs.
// ---------------------------------------------------------------------------
__global__ void initA_kernel(float* __restrict__ out, const float* __restrict__ b,
                             const float* __restrict__ x, const float* __restrict__ Wac,
                             int n) {
  int idx = blockIdx.x * blockDim.x + threadIdx.x;
  if (idx >= ((n + 1) << 4)) return;
  int i = idx >> 4, c = idx & 15;
  out[idx] = (i < n) ? (b[c] + x[i] * Wac[c]) : 0.f;
}

// Grid: (tiles over P, K)
__global__ void scatterA_kernel(const float* __restrict__ x, const float* __restrict__ Wa,
                                const int* __restrict__ rb_in, const int* __restrict__ rb_out,
                                int P, float* __restrict__ out) {
  int p = blockIdx.x * blockDim.x + threadIdx.x;
  if (p >= P) return;
  int k   = blockIdx.y;
  int idx = k * P + p;
  int r   = rb_in[idx];
  int ro  = rb_out[idx];   // padded entries target the dump row: harmless add
  float xv = x[r];
  const float* wk = Wa + k * 16;
  float* dst = out + (long)ro * 16;
#pragma unroll
  for (int c = 0; c < 16; ++c) atomicAdd(dst + c, xv * wk[c]);
}

// ---------------------------------------------------------------------------
// WMMA sparse-conv scatter: one wave32 per 16-row tile; blockIdx.y = offset k.
//   A: 16 gathered activation rows x 32ch f16  (K padded to 32 if Cin==16)
//   B: Wt[k] pre-transposed (Cout x 32) f16
//   D: 16 x COUT f32 -> global_atomic_add_f32 scatter via rb_out
// ---------------------------------------------------------------------------
template <int COUT, bool CIN16>
__global__ __launch_bounds__(256) void sconv_wmma(
    const _Float16* __restrict__ act, int actStride, int actOfs,
    const _Float16* __restrict__ Wt,
    const int* __restrict__ rb_in, const int* __restrict__ rb_out,
    int P, int n_out, float* __restrict__ out) {
  const int lane = threadIdx.x & 31;
  const int wv   = threadIdx.x >> 5;
  const int t    = blockIdx.x * (blockDim.x >> 5) + wv;   // tile within offset k
  const int tilesPerK = (P + 15) >> 4;
  if (t >= tilesPerK) return;                 // uniform per-wave: EXEC stays full
  const int k  = blockIdx.y;
  const int p0 = t << 4;
  const int lo = lane & 15;
  const int hi = lane >> 4;
  const int base = k * P;

  // prefetch next tile's rulebook entries (global_prefetch_b8)
  if (t + 1 < tilesPerK) {
    __builtin_prefetch(rb_in  + base + p0 + 16, 0, 1);
    __builtin_prefetch(rb_out + base + p0 + 16, 0, 1);
  }

  // ---- A fragment: lane holds row m=lo, K chunks [8*hi..] and [16+8*hi..]
  v16h a = {};
  {
    int pa = p0 + lo;
    if (pa < P) {
      int r = rb_in[base + pa];
      const _Float16* rp = act + (long)r * actStride + actOfs + hi * 8;
      v8h c0 = *(const v8h*)rp;
#pragma unroll
      for (int i = 0; i < 8; ++i) a[i] = c0[i];
      if (!CIN16) {
        v8h c1 = *(const v8h*)(rp + 16);
#pragma unroll
        for (int i = 0; i < 8; ++i) a[8 + i] = c1[i];
      }
    }
  }

  // ---- output rows for this lane's D elements (m = j + 8*hi)
  int rows[8];
#pragma unroll
  for (int j = 0; j < 8; ++j) {
    int pm = p0 + hi * 8 + j;
    rows[j] = (pm < P) ? rb_out[base + pm] : n_out;
  }

  const _Float16* wk = Wt + (long)k * 32 * COUT;
#pragma unroll
  for (int nt = 0; nt < COUT / 16; ++nt) {
    int n = nt * 16 + lo;
    const _Float16* bq = wk + n * 32 + hi * 8;   // K chunks [8*hi..] and [16+8*hi..]
    v8h b0 = *(const v8h*)bq;
    v8h b1 = *(const v8h*)(bq + 16);
    v16h b = __builtin_shufflevector(b0, b1, 0, 1, 2, 3, 4, 5, 6, 7, 8, 9, 10, 11, 12, 13, 14, 15);
    v8f c = {};
    v8f d = __builtin_amdgcn_wmma_f32_16x16x32_f16(
        /*neg_a=*/false, a, /*neg_b=*/false, b,
        /*c_mod=*/(short)0, c, /*reuse_a=*/false, /*reuse_b=*/false);
#pragma unroll
    for (int j = 0; j < 8; ++j)
      atomicAdd(out + (long)rows[j] * COUT + n, d[j]);
  }
}

// ---------------------------------------------------------------------------
// Layer O (16 -> 1): scalar dot + single atomic per rulebook entry.
// Grid: (tiles over P, K)
// ---------------------------------------------------------------------------
__global__ void scatterO_kernel(const _Float16* __restrict__ act, const float* __restrict__ Wo,
                                const int* __restrict__ rb_in, const int* __restrict__ rb_out,
                                int P, float* __restrict__ out) {
  int p = blockIdx.x * blockDim.x + threadIdx.x;
  if (p >= P) return;
  int k   = blockIdx.y;
  int idx = k * P + p;
  int r   = rb_in[idx];
  int ro  = rb_out[idx];
  const _Float16* row = act + (long)r * 16;
  const float* wk = Wo + k * 16;
  float s = 0.f;
#pragma unroll
  for (int j = 0; j < 16; ++j) s += (float)row[j] * wk[j];
  atomicAdd(out + ro, s);
}

// ---------------------------------------------------------------------------
// BatchNorm: sum/sumsq via LDS partial reduction + global atomics (stats pre-zeroed)
// ---------------------------------------------------------------------------
__global__ void bn_stats_kernel(const float* __restrict__ x, int n, int C,
                                float* __restrict__ stats) {
  __shared__ float ssum[32], ssq[32];
  if (threadIdx.x < 32) { ssum[threadIdx.x] = 0.f; ssq[threadIdx.x] = 0.f; }
  __syncthreads();
  int idx = blockIdx.x * blockDim.x + threadIdx.x;
  if (idx < n * C) {
    int c = idx & (C - 1);     // C is 16 or 32
    float v = x[idx];
    atomicAdd(&ssum[c], v);
    atomicAdd(&ssq[c], v * v);
  }
  __syncthreads();
  if (threadIdx.x < C) {
    atomicAdd(&stats[threadIdx.x],     ssum[threadIdx.x]);
    atomicAdd(&stats[C + threadIdx.x], ssq[threadIdx.x]);
  }
}

__global__ void bn_apply_elu_kernel(const float* __restrict__ x, int n, int cshift,
                                    const float* __restrict__ stats,
                                    const float* __restrict__ g, const float* __restrict__ bt,
                                    _Float16* __restrict__ dst, int dstStride, int dstOfs) {
  int idx = blockIdx.x * blockDim.x + threadIdx.x;
  if (idx >= (n << cshift)) return;
  int C = 1 << cshift;
  int c = idx & (C - 1);
  long i = idx >> cshift;
  float inv_n = 1.f / (float)n;
  float m  = stats[c] * inv_n;
  float vr = stats[C + c] * inv_n - m * m;
  float y  = (x[idx] - m) * rsqrtf(vr + EPS_BN) * g[c] + bt[c];
  dst[i * dstStride + dstOfs + c] = (_Float16)elu_f(y);
}

__global__ void tanh_kernel(const float* __restrict__ src, float* __restrict__ dst, int n) {
  int i = blockIdx.x * blockDim.x + threadIdx.x;
  if (i < n) dst[i] = tanhf(src[i]);
}

// ---------------------------------------------------------------------------
extern "C" void kernel_launch(void* const* d_in, const int* in_sizes, int n_in,
                              void* d_out, int out_size, void* d_ws, size_t ws_size,
                              hipStream_t stream) {
  const float* x   = (const float*)d_in[0];
  const float* Wa  = (const float*)d_in[1];
  const float* Wac = (const float*)d_in[2];
  const float* ba  = (const float*)d_in[3];
  const float* ga  = (const float*)d_in[4];
  const float* bta = (const float*)d_in[5];
  const float* Wd  = (const float*)d_in[6];
  const float* bd  = (const float*)d_in[7];
  const float* gd  = (const float*)d_in[8];
  const float* btd = (const float*)d_in[9];
  const float* Wb  = (const float*)d_in[10];
  const float* Wbc = (const float*)d_in[11];
  const float* bb  = (const float*)d_in[12];
  const float* gb  = (const float*)d_in[13];
  const float* btb = (const float*)d_in[14];
  const float* Wu  = (const float*)d_in[15];
  const float* bu  = (const float*)d_in[16];
  const float* gu  = (const float*)d_in[17];
  const float* btu = (const float*)d_in[18];
  const float* Wp  = (const float*)d_in[19];
  const float* Wpc = (const float*)d_in[20];
  const float* bp  = (const float*)d_in[21];
  const float* gp  = (const float*)d_in[22];
  const float* btp = (const float*)d_in[23];
  const float* Wo  = (const float*)d_in[24];
  const float* Woc = (const float*)d_in[25];
  const float* bo  = (const float*)d_in[26];
  const int* rbA_in  = (const int*)d_in[27];
  const int* rbA_out = (const int*)d_in[28];
  const int* rbD_in  = (const int*)d_in[29];
  const int* rbD_out = (const int*)d_in[30];
  const int* rbB_in  = (const int*)d_in[31];
  const int* rbB_out = (const int*)d_in[32];
  const int* rbU_in  = (const int*)d_in[33];
  const int* rbU_out = (const int*)d_in[34];

  const int N  = in_sizes[0];           // x is (N,1)
  const int M  = 150000;                // M_S2
  const int PA = in_sizes[27] / 26;
  const int PD = in_sizes[29] / 8;
  const int PB = in_sizes[31] / 26;
  const int PU = in_sizes[33] / 64;

  // ---- workspace carve (256B aligned) ----
  char* ws = (char*)d_ws;
  size_t off = 0;
  auto carve = [&](size_t bytes) -> void* {
    void* p = ws + off;
    off = (off + bytes + 255) & ~(size_t)255;
    return p;
  };
  float*    poolN = (float*)carve((size_t)(N + 1) * 16 * 4);  // A, U, P raw + O out
  float*    poolM = (float*)carve((size_t)(M + 1) * 32 * 4);  // D, B raw
  _Float16* d1cat = (_Float16*)carve((size_t)N * 32 * 2);     // [d1 | e1] f16
  _Float16* h16   = (_Float16*)carve((size_t)M * 32 * 2);
  _Float16* e216  = (_Float16*)carve((size_t)M * 32 * 2);
  _Float16* d1p16 = (_Float16*)carve((size_t)N * 16 * 2);
  _Float16* WdT   = (_Float16*)carve((size_t)8  * 32 * 32 * 2);
  _Float16* WbT   = (_Float16*)carve((size_t)26 * 32 * 32 * 2);
  _Float16* WuT   = (_Float16*)carve((size_t)64 * 32 * 16 * 2);
  _Float16* WpT   = (_Float16*)carve((size_t)26 * 32 * 16 * 2);
  float*    stats = (float*)carve(5 * 64 * 4);
  float *sA = stats, *sD = stats + 64, *sB = stats + 128, *sU = stats + 192, *sP = stats + 256;

  auto nblk = [](long t) { return (int)((t + 255) / 256); };
  auto wgrid = [](int P) { int tiles = (P + 15) >> 4; return (tiles + 7) / 8; };

  // ---- prep: weight transpose+f16, zero stats ----
  convert_w_kernel<<<dim3(nblk(32 * 32), 8),  256, 0, stream>>>(Wd, WdT, 16, 32);
  convert_w_kernel<<<dim3(nblk(32 * 32), 26), 256, 0, stream>>>(Wb, WbT, 32, 32);
  convert_w_kernel<<<dim3(nblk(16 * 32), 64), 256, 0, stream>>>(Wu, WuT, 32, 16);
  convert_w_kernel<<<dim3(nblk(16 * 32), 26), 256, 0, stream>>>(Wp, WpT, 32, 16);
  zero_kernel<<<nblk(320), 256, 0, stream>>>(stats, 320);

  // ---- layer A: 1 -> 16 (scalar), e1 -> d1cat channels [16,32) ----
  initA_kernel<<<nblk((long)(N + 1) * 16), 256, 0, stream>>>(poolN, ba, x, Wac, N);
  scatterA_kernel<<<dim3(nblk(PA), 26), 256, 0, stream>>>(x, Wa, rbA_in, rbA_out, PA, poolN);
  bn_stats_kernel<<<nblk((long)N * 16), 256, 0, stream>>>(poolN, N, 16, sA);
  bn_apply_elu_kernel<<<nblk((long)N * 16), 256, 0, stream>>>(poolN, N, 4, sA, ga, bta, d1cat, 32, 16);

  // ---- layer D: 16 -> 32 (WMMA, K padded), downsample rulebook ----
  init_bias_kernel<<<nblk((long)(M + 1) * 32), 256, 0, stream>>>(poolM, bd, M, 5);
  sconv_wmma<32, true><<<dim3(wgrid(PD), 8), 256, 0, stream>>>(d1cat, 32, 16, WdT, rbD_in, rbD_out, PD, M, poolM);
  bn_stats_kernel<<<nblk((long)M * 32), 256, 0, stream>>>(poolM, M, 32, sD);
  bn_apply_elu_kernel<<<nblk((long)M * 32), 256, 0, stream>>>(poolM, M, 5, sD, gd, btd, h16, 32, 0);

  // ---- layer B: 32 -> 32 (WMMA) + central Wbc ----
  init_central_kernel<<<nblk((long)(M + 1) * 32), 256, 0, stream>>>(poolM, bb, h16, 32, Wbc, M, 5);
  sconv_wmma<32, false><<<dim3(wgrid(PB), 26), 256, 0, stream>>>(h16, 32, 0, WbT, rbB_in, rbB_out, PB, M, poolM);
  bn_stats_kernel<<<nblk((long)M * 32), 256, 0, stream>>>(poolM, M, 32, sB);
  bn_apply_elu_kernel<<<nblk((long)M * 32), 256, 0, stream>>>(poolM, M, 5, sB, gb, btb, e216, 32, 0);

  // ---- layer U: 32 -> 16 (WMMA), upsample; d1 -> d1cat channels [0,16) ----
  init_bias_kernel<<<nblk((long)(N + 1) * 16), 256, 0, stream>>>(poolN, bu, N, 4);
  sconv_wmma<16, false><<<dim3(wgrid(PU), 64), 256, 0, stream>>>(e216, 32, 0, WuT, rbU_in, rbU_out, PU, N, poolN);
  bn_stats_kernel<<<nblk((long)N * 16), 256, 0, stream>>>(poolN, N, 16, sU);
  bn_apply_elu_kernel<<<nblk((long)N * 16), 256, 0, stream>>>(poolN, N, 4, sU, gu, btu, d1cat, 32, 0);

  // ---- layer P: concat(32) -> 16 (WMMA) + central Wpc ----
  init_central_kernel<<<nblk((long)(N + 1) * 16), 256, 0, stream>>>(poolN, bp, d1cat, 32, Wpc, N, 4);
  sconv_wmma<16, false><<<dim3(wgrid(PA), 26), 256, 0, stream>>>(d1cat, 32, 0, WpT, rbA_in, rbA_out, PA, N, poolN);
  bn_stats_kernel<<<nblk((long)N * 16), 256, 0, stream>>>(poolN, N, 16, sP);
  bn_apply_elu_kernel<<<nblk((long)N * 16), 256, 0, stream>>>(poolN, N, 4, sP, gp, btp, d1p16, 16, 0);

  // ---- layer O: 16 -> 1 (scalar) + central Woc, then tanh ----
  init_central_kernel<<<nblk((long)(N + 1)), 256, 0, stream>>>(poolN, bo, d1p16, 16, Woc, N, 0);
  scatterO_kernel<<<dim3(nblk(PA), 26), 256, 0, stream>>>(d1p16, Wo, rbA_in, rbA_out, PA, poolN);
  tanh_kernel<<<nblk(N), 256, 0, stream>>>(poolN, (float*)d_out, N);
}
